// NodeClassificationMPNSimple_59072980189453
// MI455X (gfx1250) — matrix operand order repaired and checked by reading, not compiled, not run on hardware
//
#include <hip/hip_runtime.h>
#include <cstdint>
#include <cstddef>

typedef __bf16 bf16_t;
typedef __attribute__((ext_vector_type(16))) __bf16 v16bf;
typedef __attribute__((ext_vector_type(8)))  float  v8f;

#define NNODES 50000
#define NEDGES 800000
#define NODE_TILES (NNODES / 16)  // 3125 (exact)
#define EDGE_TILES (NEDGES / 16)  // 50000 (exact)

#define EMLP_WAVES 8
#define EMLP_BLOCKS 625           // 625*8 = 5000 wave slots -> exactly 10 tiles/wave
#define EEMB_WAVES 8
#define EEMB_BLOCKS 625
#define NODE_WAVES 8
#define NODE_BLOCKS 40            // 320 wave slots over 3125 tiles (~10 tiles/wave)

// ---------------------------------------------------------------------------
// WMMA helpers (CDNA5 gfx1250, wave32)
// ---------------------------------------------------------------------------
__device__ __forceinline__ v8f wmma_bf16(v16bf a, v16bf b, v8f c) {
  return __builtin_amdgcn_wmma_f32_16x16x32_bf16(
      /*neg_a=*/false, a, /*neg_b=*/false, b,
      /*c_mod=*/(short)0, c, /*reuse_a=*/false, /*reuse_b=*/false);
}

// A fragment (16x32 bf16): lane<16 -> row M=lane, K={kb..kb+7, kb+16..kb+23};
// lane>=16 -> same row set M=lane-16, K shifted +8.
__device__ __forceinline__ v16bf fragA_bf16(const bf16_t* row, int kbase, int lane) {
  const int k0 = kbase + ((lane & 16) >> 1);
  v16bf f;
#pragma unroll
  for (int j = 0; j < 8; ++j) {
    f[j]     = row[k0 + j];
    f[j + 8] = row[k0 + 16 + j];
  }
  return f;
}

__device__ __forceinline__ v16bf fragA_f32(const float* row, int kbase, int lane) {
  const int k0 = kbase + ((lane & 16) >> 1);
  v16bf f;
#pragma unroll
  for (int j = 0; j < 8; ++j) {
    f[j]     = (bf16_t)row[k0 + j];
    f[j + 8] = (bf16_t)row[k0 + 16 + j];
  }
  return f;
}

__device__ __forceinline__ v16bf fragA_f32_bounded(const float* row, int kbase, int lane, int Kmax) {
  const int k0 = kbase + ((lane & 16) >> 1);
  v16bf f;
#pragma unroll
  for (int j = 0; j < 8; ++j) {
    const int ka = k0 + j, kb = k0 + 16 + j;
    f[j]     = (ka < Kmax) ? (bf16_t)row[ka] : (bf16_t)0.0f;
    f[j + 8] = (kb < Kmax) ? (bf16_t)row[kb] : (bf16_t)0.0f;
  }
  return f;
}

// B fragment (32x16 bf16): lane<16 -> col N=lane, K={kb..kb+7, kb+16..kb+23}; lane>=16 -> K shifted +8.
// colseg = column stored contiguously in K (column-major weights in LDS).
__device__ __forceinline__ v16bf fragB_bf16(const bf16_t* colseg, int kbase, int lane) {
  const int k0 = kbase + ((lane & 16) >> 1);
  v16bf f;
#pragma unroll
  for (int j = 0; j < 8; ++j) {
    f[j]     = colseg[k0 + j];
    f[j + 8] = colseg[k0 + 16 + j];
  }
  return f;
}

// Stage W[K][Nout] (row-major f32) -> LDS column-major bf16 [Nout][Kpad]
__device__ __forceinline__ void stage_w_pad(bf16_t* dst, const float* w, int K, int Kpad,
                                            int Nout, int tid, int nth) {
  for (int t = tid; t < Nout * Kpad; t += nth) {
    const int n = t / Kpad, k = t % Kpad;
    dst[t] = (k < K) ? (bf16_t)w[k * Nout + n] : (bf16_t)0.0f;
  }
}

__device__ __forceinline__ void lds_fence() {
  asm volatile("s_wait_dscnt 0" ::: "memory");
}

// ---------------------------------------------------------------------------
__global__ void zero_kernel(float* __restrict__ p, int n) {
  const int i = blockIdx.x * blockDim.x + threadIdx.x;
  if (i < n) p[i] = 0.0f;
}

// ---------------------------------------------------------------------------
// Node embedding: nf0 = relu(x @ W1 + b1) @ W2 + b2
// writes nfc[:,0:64] = nfc[:,64:128] = bf16(nf0).  x:[N,128] W1:[128,64] W2:[64,64]
// ---------------------------------------------------------------------------
__global__ void __launch_bounds__(32 * NODE_WAVES)
node_embed_kernel(const float* __restrict__ x,
                  const float* __restrict__ W1, const float* __restrict__ b1,
                  const float* __restrict__ W2, const float* __restrict__ b2,
                  bf16_t* __restrict__ nfc) {
  __shared__ bf16_t w1[64 * 128];
  __shared__ bf16_t w2[64 * 64];
  __shared__ bf16_t h[NODE_WAVES][16][64];
  const int tid = threadIdx.x, lane = tid & 31, wave = tid >> 5;
  stage_w_pad(w1, W1, 128, 128, 64, tid, blockDim.x);
  stage_w_pad(w2, W2, 64, 64, 64, tid, blockDim.x);
  __syncthreads();

  const int m = lane & 15;
  const int mrow = (lane & 16) >> 1;
  const int stride = gridDim.x * NODE_WAVES;

  for (int tile = blockIdx.x * NODE_WAVES + wave; tile < NODE_TILES; tile += stride) {
    const float* arow = x + (size_t)(tile * 16 + m) * 128;

    v8f acc[4] = {};
#pragma unroll
    for (int c = 0; c < 4; ++c) {
      const v16bf a = fragA_f32(arow, c * 32, lane);
#pragma unroll
      for (int t = 0; t < 4; ++t) {
        const v16bf b = fragB_bf16(&w1[(t * 16 + m) * 128], c * 32, lane);
        acc[t] = wmma_bf16(a, b, acc[t]);
      }
    }
#pragma unroll
    for (int t = 0; t < 4; ++t) {
      const float bias = b1[t * 16 + m];
#pragma unroll
      for (int i = 0; i < 8; ++i) {
        const float v = acc[t][i] + bias;
        h[wave][mrow + i][t * 16 + m] = (bf16_t)(v > 0.0f ? v : 0.0f);
      }
    }
    lds_fence();

    v8f o[4] = {};
#pragma unroll
    for (int c = 0; c < 2; ++c) {
      const v16bf a = fragA_bf16(&h[wave][m][0], c * 32, lane);
#pragma unroll
      for (int t = 0; t < 4; ++t) {
        const v16bf b = fragB_bf16(&w2[(t * 16 + m) * 64], c * 32, lane);
        o[t] = wmma_bf16(a, b, o[t]);
      }
    }
    const int rowbase = tile * 16 + mrow;
#pragma unroll
    for (int t = 0; t < 4; ++t) {
      const int n = t * 16 + m;
      const float bias = b2[n];
#pragma unroll
      for (int i = 0; i < 8; ++i) {
        const bf16_t bv = (bf16_t)(o[t][i] + bias);
        bf16_t* row = nfc + (size_t)(rowbase + i) * 128;
        row[n] = bv;
        row[64 + n] = bv;
      }
    }
  }
}

// ---------------------------------------------------------------------------
// Edge embedding: ef0 = relu(ea @ W1 + b1) @ W2 + b2
// writes efc[:,0:64] = efc[:,64:128] = bf16(ef0).  ea:[E,19] (K pad 32)
// ---------------------------------------------------------------------------
__global__ void __launch_bounds__(32 * EEMB_WAVES)
edge_embed_kernel(const float* __restrict__ ea,
                  const float* __restrict__ W1, const float* __restrict__ b1,
                  const float* __restrict__ W2, const float* __restrict__ b2,
                  bf16_t* __restrict__ efc) {
  __shared__ bf16_t w1[64 * 32];
  __shared__ bf16_t w2[64 * 64];
  __shared__ bf16_t h[EEMB_WAVES][16][64];
  const int tid = threadIdx.x, lane = tid & 31, wave = tid >> 5;
  stage_w_pad(w1, W1, 19, 32, 64, tid, blockDim.x);
  stage_w_pad(w2, W2, 64, 64, 64, tid, blockDim.x);
  __syncthreads();

  const int m = lane & 15;
  const int mrow = (lane & 16) >> 1;
  const int stride = gridDim.x * EEMB_WAVES;

  for (int tile = blockIdx.x * EEMB_WAVES + wave; tile < EDGE_TILES; tile += stride) {
    const float* arow = ea + (size_t)(tile * 16 + m) * 19;

    v8f acc[4] = {};
    const v16bf a0 = fragA_f32_bounded(arow, 0, lane, 19);
#pragma unroll
    for (int t = 0; t < 4; ++t) {
      const v16bf b = fragB_bf16(&w1[(t * 16 + m) * 32], 0, lane);
      acc[t] = wmma_bf16(a0, b, acc[t]);
    }
#pragma unroll
    for (int t = 0; t < 4; ++t) {
      const float bias = b1[t * 16 + m];
#pragma unroll
      for (int i = 0; i < 8; ++i) {
        const float v = acc[t][i] + bias;
        h[wave][mrow + i][t * 16 + m] = (bf16_t)(v > 0.0f ? v : 0.0f);
      }
    }
    lds_fence();

    v8f o[4] = {};
#pragma unroll
    for (int c = 0; c < 2; ++c) {
      const v16bf a = fragA_bf16(&h[wave][m][0], c * 32, lane);
#pragma unroll
      for (int t = 0; t < 4; ++t) {
        const v16bf b = fragB_bf16(&w2[(t * 16 + m) * 64], c * 32, lane);
        o[t] = wmma_bf16(a, b, o[t]);
      }
    }
    const int rowbase = tile * 16 + mrow;
#pragma unroll
    for (int t = 0; t < 4; ++t) {
      const int n = t * 16 + m;
      const float bias = b2[n];
#pragma unroll
      for (int i = 0; i < 8; ++i) {
        const bf16_t bv = (bf16_t)(o[t][i] + bias);
        bf16_t* row = efc + (size_t)(rowbase + i) * 128;
        row[n] = bv;
        row[64 + n] = bv;
      }
    }
  }
}

// ---------------------------------------------------------------------------
// Edge MLP step (hot kernel):
//  msg = [nfc[src] | nfc[dst] | efc[e]] (K=384 bf16 gathers, single pass)
//  e_new = relu(msg @ W1 + b1) @ W2 + b2 -> efc[:,64:128]; agg[dst] += e_new
// Full W1 (384x128, 96 KB) + W2 (128x64, 16 KB) staged ONCE per block in
// dynamic LDS (320 KB/WGP on CDNA5); persistent tile loop amortizes staging;
// global_prefetch_b8 hides the random-gather latency one grid-stride ahead.
// ---------------------------------------------------------------------------
__global__ void __launch_bounds__(32 * EMLP_WAVES)
edge_mlp_kernel(const bf16_t* __restrict__ nfc,
                bf16_t* __restrict__ efc,
                const int* __restrict__ eidx,   // [2,E]
                const float* __restrict__ W1, const float* __restrict__ b1,
                const float* __restrict__ W2, const float* __restrict__ b2,
                float* __restrict__ agg) {
  extern __shared__ __align__(16) char smem[];
  bf16_t* w1 = (bf16_t*)smem;               // [128 cols][384 K]   96 KB
  bf16_t* w2 = w1 + 128 * 384;              // [64 cols][128 K]    16 KB
  bf16_t* hb = w2 + 64 * 128;               // [EMLP_WAVES][16][128] 32 KB

  const int tid = threadIdx.x, lane = tid & 31, wave = tid >> 5;
  // stage full W1 column-major bf16
  for (int t = tid; t < 128 * 384; t += blockDim.x) {
    const int n = t / 384, k = t % 384;
    w1[t] = (bf16_t)W1[k * 128 + n];
  }
  stage_w_pad(w2, W2, 128, 128, 64, tid, blockDim.x);
  __syncthreads();

  const int m = lane & 15;
  const int mrow = (lane & 16) >> 1;
  const int stride = gridDim.x * EMLP_WAVES;
  bf16_t* hrow = hb + (wave * 16 + 0) * 128;

  for (int tile = blockIdx.x * EMLP_WAVES + wave; tile < EDGE_TILES; tile += stride) {
    const int e = tile * 16 + m;
    const bf16_t* segs[3];
    segs[0] = nfc + (size_t)eidx[e] * 128;            // x_src (nf0|nf)
    segs[1] = nfc + (size_t)eidx[NEDGES + e] * 128;   // x_dst
    segs[2] = efc + (size_t)e * 128;                  // (ef0|ef)

    // prefetch next tile's gather rows (one grid-stride ahead)
    const int tnext = tile + stride;
    if (tnext < EDGE_TILES) {
      const int en = tnext * 16 + m;
      const bf16_t* ps = nfc + (size_t)eidx[en] * 128;
      const bf16_t* pd = nfc + (size_t)eidx[NEDGES + en] * 128;
      __builtin_prefetch(ps, 0, 1);
      __builtin_prefetch(ps + 64, 0, 1);
      __builtin_prefetch(pd, 0, 1);
      __builtin_prefetch(pd + 64, 0, 1);
    }

    // ---- layer 1: [16,384] @ [384,128], all 8 N-tiles in one pass ----
    v8f acc[8] = {};
#pragma unroll 1
    for (int c = 0; c < 12; ++c) {
      const v16bf a = fragA_bf16(segs[c >> 2], (c & 3) * 32, lane);
#pragma unroll
      for (int t = 0; t < 8; ++t) {
        const v16bf b = fragB_bf16(&w1[(t * 16 + m) * 384], c * 32, lane);
        acc[t] = wmma_bf16(a, b, acc[t]);
      }
    }
#pragma unroll
    for (int t = 0; t < 8; ++t) {
      const int n = t * 16 + m;
      const float bias = b1[n];
#pragma unroll
      for (int i = 0; i < 8; ++i) {
        const float v = acc[t][i] + bias;
        hb[(wave * 16 + mrow + i) * 128 + n] = (bf16_t)(v > 0.0f ? v : 0.0f);
      }
    }
    lds_fence();

    // ---- layer 2: [16,128] @ [128,64] ----
    v8f o[4] = {};
#pragma unroll
    for (int c = 0; c < 4; ++c) {
      const v16bf a = fragA_bf16(hrow + m * 128, c * 32, lane);
#pragma unroll
      for (int t = 0; t < 4; ++t) {
        const v16bf b = fragB_bf16(&w2[(t * 16 + m) * 128], c * 32, lane);
        o[t] = wmma_bf16(a, b, o[t]);
      }
    }

    // epilogue: write e_new (bf16) and scatter-add into agg
    int d8[8];
#pragma unroll
    for (int i = 0; i < 8; ++i) d8[i] = eidx[NEDGES + tile * 16 + mrow + i];
#pragma unroll
    for (int t = 0; t < 4; ++t) {
      const int n = t * 16 + m;
      const float bias = b2[n];
#pragma unroll
      for (int i = 0; i < 8; ++i) {
        const float v = o[t][i] + bias;
        const int er = tile * 16 + mrow + i;
        efc[(size_t)er * 128 + 64 + n] = (bf16_t)v;
        atomicAdd(agg + (size_t)d8[i] * 64 + n, v);
      }
    }
  }
}

// ---------------------------------------------------------------------------
// Node update: nf = relu(agg @ W + b); nfc[:,64:128] = bf16(nf); agg tile := 0
// ---------------------------------------------------------------------------
__global__ void __launch_bounds__(32 * NODE_WAVES)
node_update_kernel(float* __restrict__ agg,
                   const float* __restrict__ W, const float* __restrict__ b,
                   float* __restrict__ nf, bf16_t* __restrict__ nfc) {
  __shared__ bf16_t w[64 * 64];
  const int tid = threadIdx.x, lane = tid & 31, wave = tid >> 5;
  stage_w_pad(w, W, 64, 64, 64, tid, blockDim.x);
  __syncthreads();

  const int m = lane & 15;
  const int mrow = (lane & 16) >> 1;
  const int stride = gridDim.x * NODE_WAVES;

  for (int tile = blockIdx.x * NODE_WAVES + wave; tile < NODE_TILES; tile += stride) {
    const float* arow = agg + (size_t)(tile * 16 + m) * 64;

    v8f acc[4] = {};
#pragma unroll
    for (int c = 0; c < 2; ++c) {
      const v16bf a = fragA_f32(arow, c * 32, lane);
#pragma unroll
      for (int t = 0; t < 4; ++t) {
        const v16bf bb = fragB_bf16(&w[(t * 16 + m) * 64], c * 32, lane);
        acc[t] = wmma_bf16(a, bb, acc[t]);
      }
    }
    const int rowbase = tile * 16 + mrow;
#pragma unroll
    for (int t = 0; t < 4; ++t) {
      const int n = t * 16 + m;
      const float bias = b[n];
#pragma unroll
      for (int i = 0; i < 8; ++i) {
        float v = acc[t][i] + bias;
        v = v > 0.0f ? v : 0.0f;
        nf[(size_t)(rowbase + i) * 64 + n] = v;
        nfc[(size_t)(rowbase + i) * 128 + 64 + n] = (bf16_t)v;
      }
    }
    // re-zero this tile's agg rows for the next step
    float* zb = agg + (size_t)tile * 16 * 64;
    for (int idx = lane; idx < 16 * 64; idx += 32) zb[idx] = 0.0f;
  }
}

// ---------------------------------------------------------------------------
// Edge head: pred_edge = relu(ef @ W1 + b1) @ W2 + b2   (64->32->1)
// ---------------------------------------------------------------------------
__global__ void __launch_bounds__(256)
edge_head_kernel(const bf16_t* __restrict__ efc,
                 const float* __restrict__ W1, const float* __restrict__ b1,
                 const float* __restrict__ W2, const float* __restrict__ b2,
                 float* __restrict__ out) {
  __shared__ float w1[64 * 32];
  __shared__ float w2s[32];
  for (int t = threadIdx.x; t < 64 * 32; t += blockDim.x) w1[t] = W1[t];
  if (threadIdx.x < 32) w2s[threadIdx.x] = W2[threadIdx.x];
  __syncthreads();
  const int e = blockIdx.x * blockDim.x + threadIdx.x;
  if (e >= NEDGES) return;
  float f[64];
  const bf16_t* row = efc + (size_t)e * 128 + 64;
#pragma unroll
  for (int k = 0; k < 64; ++k) f[k] = (float)row[k];
  float acc = b2[0];
#pragma unroll 4
  for (int j = 0; j < 32; ++j) {
    float hh = b1[j];
#pragma unroll
    for (int k = 0; k < 64; ++k) hh = fmaf(f[k], w1[k * 32 + j], hh);
    acc = fmaf(fmaxf(hh, 0.0f), w2s[j], acc);
  }
  out[e] = acc;
}

// ---------------------------------------------------------------------------
// Node heads: pred_node (64->32->1), pred_class (64->32->17)
// ---------------------------------------------------------------------------
__global__ void __launch_bounds__(256)
node_head_kernel(const float* __restrict__ nf,
                 const float* __restrict__ ncW1, const float* __restrict__ ncb1,
                 const float* __restrict__ ncW2, const float* __restrict__ ncb2,
                 const float* __restrict__ clW1, const float* __restrict__ clb1,
                 const float* __restrict__ clW2, const float* __restrict__ clb2,
                 float* __restrict__ out_node, float* __restrict__ out_class) {
  __shared__ float wn1[64 * 32];
  __shared__ float wc1[64 * 32];
  __shared__ float wn2[32];
  __shared__ float wc2[32 * 17];
  for (int t = threadIdx.x; t < 64 * 32; t += blockDim.x) { wn1[t] = ncW1[t]; wc1[t] = clW1[t]; }
  for (int t = threadIdx.x; t < 32 * 17; t += blockDim.x) wc2[t] = clW2[t];
  if (threadIdx.x < 32) wn2[threadIdx.x] = ncW2[threadIdx.x];
  __syncthreads();
  const int i = blockIdx.x * blockDim.x + threadIdx.x;
  if (i >= NNODES) return;
  float f[64];
  const float* row = nf + (size_t)i * 64;
#pragma unroll
  for (int k = 0; k < 64; ++k) f[k] = row[k];

  float accn = ncb2[0];
  float cls[17];
#pragma unroll
  for (int c = 0; c < 17; ++c) cls[c] = clb2[c];
#pragma unroll 2
  for (int j = 0; j < 32; ++j) {
    float hn = ncb1[j], hc = clb1[j];
#pragma unroll
    for (int k = 0; k < 64; ++k) {
      hn = fmaf(f[k], wn1[k * 32 + j], hn);
      hc = fmaf(f[k], wc1[k * 32 + j], hc);
    }
    accn = fmaf(fmaxf(hn, 0.0f), wn2[j], accn);
    const float hcr = fmaxf(hc, 0.0f);
#pragma unroll
    for (int c = 0; c < 17; ++c) cls[c] = fmaf(hcr, wc2[j * 17 + c], cls[c]);
  }
  out_node[i] = accn;
#pragma unroll
  for (int c = 0; c < 17; ++c) out_class[(size_t)i * 17 + c] = cls[c];
}

// ---------------------------------------------------------------------------
// Host launcher
// ---------------------------------------------------------------------------
static inline size_t align256(size_t x) { return (x + 255) & ~(size_t)255; }

extern "C" void kernel_launch(void* const* d_in, const int* in_sizes, int n_in,
                              void* d_out, int out_size, void* d_ws, size_t ws_size,
                              hipStream_t stream) {
  (void)in_sizes; (void)n_in; (void)out_size; (void)ws_size;
  const float* x    = (const float*)d_in[0];
  const float* ea   = (const float*)d_in[1];
  const int*   eidx = (const int*)  d_in[2];
  // d_in[3] = node_types (unused by reference)
  const float* neW1 = (const float*)d_in[4],  *neb1 = (const float*)d_in[5];
  const float* neW2 = (const float*)d_in[6],  *neb2 = (const float*)d_in[7];
  const float* eeW1 = (const float*)d_in[8],  *eeb1 = (const float*)d_in[9];
  const float* eeW2 = (const float*)d_in[10], *eeb2 = (const float*)d_in[11];
  const float* mpW1 = (const float*)d_in[12], *mpb1 = (const float*)d_in[13];
  const float* mpW2 = (const float*)d_in[14], *mpb2 = (const float*)d_in[15];
  const float* mnW  = (const float*)d_in[16], *mnb  = (const float*)d_in[17];
  const float* ecW1 = (const float*)d_in[18], *ecb1 = (const float*)d_in[19];
  const float* ecW2 = (const float*)d_in[20], *ecb2 = (const float*)d_in[21];
  const float* ncW1 = (const float*)d_in[22], *ncb1 = (const float*)d_in[23];
  const float* ncW2 = (const float*)d_in[24], *ncb2 = (const float*)d_in[25];
  const float* clW1 = (const float*)d_in[26], *clb1 = (const float*)d_in[27];
  const float* clW2 = (const float*)d_in[28], *clb2 = (const float*)d_in[29];

  float* out = (float*)d_out;
  float* out_edge  = out;                    // [E]
  float* out_node  = out + NEDGES;           // [N]
  float* out_class = out + NEDGES + NNODES;  // [N,17]

  // workspace carve
  char* ws = (char*)d_ws;
  bf16_t* nfc = (bf16_t*)ws; ws += align256((size_t)NNODES * 128 * sizeof(bf16_t));
  bf16_t* efc = (bf16_t*)ws; ws += align256((size_t)NEDGES * 128 * sizeof(bf16_t));
  float*  agg = (float*) ws; ws += align256((size_t)NNODES * 64 * sizeof(float));
  float*  nf  = (float*) ws; ws += align256((size_t)NNODES * 64 * sizeof(float));

  // dynamic LDS for edge_mlp: W1cm 96KB + W2cm 16KB + hidden 32KB = 144KB/block
  const size_t emlp_lds = (size_t)(128 * 384 + 64 * 128 + EMLP_WAVES * 16 * 128) * sizeof(bf16_t);

  zero_kernel<<<(NNODES * 64 + 255) / 256, 256, 0, stream>>>(agg, NNODES * 64);
  node_embed_kernel<<<NODE_BLOCKS, 32 * NODE_WAVES, 0, stream>>>(x, neW1, neb1, neW2, neb2, nfc);
  edge_embed_kernel<<<EEMB_BLOCKS, 32 * EEMB_WAVES, 0, stream>>>(ea, eeW1, eeb1, eeW2, eeb2, efc);

  for (int s = 0; s < 6; ++s) {
    edge_mlp_kernel<<<EMLP_BLOCKS, 32 * EMLP_WAVES, emlp_lds, stream>>>(
        nfc, efc, eidx, mpW1, mpb1, mpW2, mpb2, agg);
    node_update_kernel<<<NODE_BLOCKS, 32 * NODE_WAVES, 0, stream>>>(agg, mnW, mnb, nf, nfc);
    if (s == 3) {
      edge_head_kernel<<<(NEDGES + 255) / 256, 256, 0, stream>>>(
          efc, ecW1, ecb1, ecW2, ecb2, out_edge);
    }
  }
  node_head_kernel<<<(NNODES + 255) / 256, 256, 0, stream>>>(
      nf, ncW1, ncb1, ncW2, ncb2, clW1, clb1, clW2, clb2, out_node, out_class);
}
